// GroupedExperts_3539053051928
// MI455X (gfx1250) — matrix operand order repaired and checked by reading, not compiled, not run on hardware
//
#include <hip/hip_runtime.h>

// ---------------- problem constants (match reference) ----------------
#define TOTAL_TOKENS 65536
#define DIM          1024
#define NEXP         8
#define MEL          (DIM * DIM)        // elements per expert matrix
// ---------------- tiling ----------------
#define BM   128           // tokens per block
#define BN   128           // hidden/out chunk width
#define KB   32            // K depth per stage (one bf16 WMMA K)
#define NCH  (DIM / BN)    // 8 chunks
#define NKB  (DIM / KB)    // 32 k-steps
#define WROW (KB + 8)      // padded LDS row: 64B data + 16B pad (TDM pad reproduces this)
#define HROW (DIM + 8)     // padded LDS row for H (1032 u16 = 2064B)

typedef __bf16 bf16x8  __attribute__((ext_vector_type(8)));
typedef __bf16 bf16x16 __attribute__((ext_vector_type(16)));
typedef float  f32x8   __attribute__((ext_vector_type(8)));
typedef float  f32x4   __attribute__((ext_vector_type(4)));
typedef unsigned short u16x4 __attribute__((ext_vector_type(4)));
typedef unsigned short u16x8 __attribute__((ext_vector_type(8)));
typedef unsigned int   u32x4v __attribute__((ext_vector_type(4)));
typedef int            i32x8v __attribute__((ext_vector_type(8)));
typedef int            i32x4v __attribute__((ext_vector_type(4)));

static __device__ __forceinline__ unsigned short f2bf(float f) {
    union { float f; unsigned int u; } v; v.f = f;
    unsigned int u = v.u;
    unsigned int r = u + 0x7fffu + ((u >> 16) & 1u);   // round-to-nearest-even
    return (unsigned short)(r >> 16);
}

static __device__ __forceinline__ float fast_rcp(float x) {
#if __has_builtin(__builtin_amdgcn_rcpf)
    return __builtin_amdgcn_rcpf(x);     // v_rcp_f32
#else
    return 1.0f / x;
#endif
}

// A fragment (16x32 bf16, row-major in LDS, contiguous K per row).
static __device__ __forceinline__ bf16x16 frag_a(const unsigned short* rowp, int hhi) {
    const bf16x8 lo = *(const bf16x8*)(rowp + hhi * 16);
    const bf16x8 hi = *(const bf16x8*)(rowp + hhi * 16 + 8);
    return __builtin_shufflevector(lo, hi, 0,1,2,3,4,5,6,7,8,9,10,11,12,13,14,15);
}

// B fragment (32x16 bf16) from K-contiguous LDS tile WT[n][k]: one 32B read.
static __device__ __forceinline__ bf16x16 frag_b(const unsigned short* p) {
    return *(const bf16x16*)p;
}

static __device__ __forceinline__ f32x8 wmma_bf16(bf16x16 a, bf16x16 b, f32x8 c) {
    return __builtin_amdgcn_wmma_f32_16x16x32_bf16(
        false, a, false, b, (short)0, c, false, false);
}

// ---------------------------------------------------------------------------
// TDM: DMA one BN(=128 rows) x KB(=32 bf16) tile of a pre-transposed bf16
// weight matrix [N][K] into LDS laid out as WT[n][WROW] (64B data + 16B pad).
// D# per CDNA5 ISA (group0/1; groups 2,3 zero for 2-D).
// clang-23 toolchain: 6-arg builtin (g0, g1, g2, g3, extra, cpol).
// ---------------------------------------------------------------------------
static __device__ __forceinline__ void tdm_load_wtile(const unsigned short* gtile,
                                                      const unsigned short* ldsdst) {
    const unsigned long long ga = (unsigned long long)(size_t)gtile;
    const unsigned lds_off = (unsigned)(size_t)ldsdst;
    u32x4v g0;
    g0[0] = 1u;                                              // count=1 (valid user D#)
    g0[1] = lds_off;                                         // lds_addr
    g0[2] = (unsigned)ga;                                    // global_addr[31:0]
    g0[3] = (unsigned)((ga >> 32) & 0x01ffffffu) | (2u << 30); // addr[56:32] | type=2
    i32x8v g1;
    // data_size=1 (2B) | pad_enable | pad_interval=3 (16 DW = 64B) | pad_amount=3 (4 DW = 16B)
    g1[0] = (int)((1u << 16) | (1u << 20) | (3u << 22) | (3u << 25));
    g1[1] = (int)(1024u << 16);        // tensor_dim0[15:0]  (=1024, >= tile)
    g1[2] = (int)(1024u << 16);        // tensor_dim0 hi=0 | tensor_dim1[15:0]=1024
    g1[3] = (int)(32u << 16);          // tensor_dim1 hi=0 | tile_dim0 = 32 (k, contiguous)
    g1[4] = (int)128u;                 // tile_dim1 = 128 (n rows) | tile_dim2 = 0
    g1[5] = (int)1024u;                // tensor_dim0_stride[31:0] = 1024 elements
    g1[6] = 0;                         // stride0 hi | stride1 lo
    g1[7] = 0;
    const i32x4v gz4 = {0, 0, 0, 0};
    const i32x8v gz8 = {0, 0, 0, 0, 0, 0, 0, 0};
    __builtin_amdgcn_tensor_load_to_lds(g0, g1, gz4, gz4, gz8, 0);
}

// ---------------------------------------------------------------------------
// Pre-pass: convert f32 weights to bf16 AND transpose to [n][k], via LDS tiles.
// Grid: (16, 16, 24)  z = mat*8 + expert ; block 256.
// ---------------------------------------------------------------------------
extern "C" __global__ __launch_bounds__(256)
void moe_wcvt_kernel(const float* __restrict__ w1,
                     const float* __restrict__ w2,
                     const float* __restrict__ w3,
                     unsigned short* __restrict__ wsT) {
    __shared__ unsigned short T[64][72];
    const int z   = blockIdx.z;
    const int mat = z >> 3;
    const int e   = z & 7;
    const float* src = (mat == 0 ? w1 : (mat == 1 ? w3 : w2)) + (size_t)e * MEL;
    unsigned short* dst = wsT + (size_t)z * MEL;
    const int kb0 = blockIdx.x * 64;
    const int nb0 = blockIdx.y * 64;
    const int tid = threadIdx.x;

    #pragma unroll
    for (int i = 0; i < 4; ++i) {                 // read 64x64 f32 tile, coalesced
        int c  = tid + i * 256;
        int kk = c >> 4;
        int n4 = (c & 15) * 4;
        const f32x4 v = *(const f32x4*)(src + (size_t)(kb0 + kk) * DIM + nb0 + n4);
        T[n4 + 0][kk] = f2bf(v[0]);
        T[n4 + 1][kk] = f2bf(v[1]);
        T[n4 + 2][kk] = f2bf(v[2]);
        T[n4 + 3][kk] = f2bf(v[3]);
    }
    __syncthreads();
    #pragma unroll
    for (int i = 0; i < 2; ++i) {                 // write 64x64 bf16 transposed, coalesced
        int c  = tid + i * 256;
        int r  = c >> 3;
        int k8 = (c & 7) * 8;
        *(u16x8*)(dst + (size_t)(nb0 + r) * DIM + kb0 + k8) = *(const u16x8*)&T[r][k8];
    }
}

// ---------------------------------------------------------------------------
// Fused grouped FFN: out = (silu(x@w1) * (x@w3)) @ w2 per expert.
// TDM double-buffered weight tiles, one barrier per k-step.
// ---------------------------------------------------------------------------
extern "C" __global__ __launch_bounds__(256)
void moe_ffn_fused_kernel(const float* __restrict__ x,
                          const unsigned short* __restrict__ w1t,
                          const unsigned short* __restrict__ w3t,
                          const unsigned short* __restrict__ w2t,
                          const long long* __restrict__ m_sizes,
                          float* __restrict__ out) {
    __shared__ unsigned short Hs[BM][HROW];       // 258 KB : full hidden rows (bf16)
    __shared__ unsigned short Xs[2][BM][WROW];    // 20 KB  : double-buffered X tiles
    __shared__ unsigned short WA[2][BN][WROW];    // 20 KB  : w1 (phase A) / w2 (phase B)
    __shared__ unsigned short WB[2][BN][WROW];    // 20 KB  : w3 (phase A)

    const int e = blockIdx.y;
    long long off = 0;
    #pragma unroll
    for (int i = 0; i < NEXP; ++i) off += (i < e) ? m_sizes[i] : 0;
    const long long rows = m_sizes[e];
    const long long row0 = (long long)blockIdx.x * BM;
    if (row0 >= rows) return;

    const float*          xe   = x   + (size_t)off * DIM;
    const unsigned short* w1te = w1t + (size_t)e * MEL;
    const unsigned short* w3te = w3t + (size_t)e * MEL;
    const unsigned short* w2te = w2t + (size_t)e * MEL;
    float*                oe   = out + (size_t)off * DIM;

    const int tid   = threadIdx.x;
    const int lane  = tid & 31;
    const int wave  = tid >> 5;       // 0..7 -> 16 token-rows each
    const int lhalf = lane & 15;
    const int hhi   = lane >> 4;

    // stage one 128x32 X tile (f32 -> bf16) into Xs[buf]
    auto stage_x = [&](int kb, int buf) {
        #pragma unroll
        for (int i = 0; i < 4; ++i) {
            int c  = tid + i * 256;
            int r  = c >> 3;
            int c4 = (c & 7) * 4;
            long long gr = row0 + r; if (gr >= rows) gr = rows - 1;
            const f32x4 v = *(const f32x4*)(xe + (size_t)gr * DIM + kb * KB + c4);
            u16x4 s; s[0] = f2bf(v[0]); s[1] = f2bf(v[1]); s[2] = f2bf(v[2]); s[3] = f2bf(v[3]);
            *(u16x4*)&Xs[buf][r][c4] = s;
        }
    };

    // ================= Phase A: H = silu(X*W1) .* (X*W3) ==================
    for (int nc = 0; nc < NCH; ++nc) {
        f32x8 acc1[8] = {};
        f32x8 acc3[8] = {};
        const unsigned short* w1nc = w1te + (size_t)(nc * BN) * DIM;
        const unsigned short* w3nc = w3te + (size_t)(nc * BN) * DIM;

        if (wave == 0) {
            tdm_load_wtile(w1nc, &WA[0][0][0]);
            tdm_load_wtile(w3nc, &WB[0][0][0]);
        }
        stage_x(0, 0);
        if (wave == 0) __builtin_amdgcn_s_wait_tensorcnt(0);

        for (int kb = 0; kb < NKB; ++kb) {
            const int cur = kb & 1, nxt = cur ^ 1;
            __syncthreads();
            if (kb + 1 < NKB) {
                if (wave == 0) {
                    tdm_load_wtile(w1nc + (kb + 1) * KB, &WA[nxt][0][0]);
                    tdm_load_wtile(w3nc + (kb + 1) * KB, &WB[nxt][0][0]);
                }
                stage_x(kb + 1, nxt);
            }
            const bf16x16 a = frag_a(&Xs[cur][wave * 16 + lhalf][0], hhi);
            #pragma unroll
            for (int nt = 0; nt < 8; ++nt) {
                const int n = nt * 16 + lhalf;
                acc1[nt] = wmma_bf16(a, frag_b(&WA[cur][n][hhi * 16]), acc1[nt]);
                acc3[nt] = wmma_bf16(a, frag_b(&WB[cur][n][hhi * 16]), acc3[nt]);
            }
            if (kb + 1 < NKB && wave == 0) __builtin_amdgcn_s_wait_tensorcnt(0);
        }
        // epilogue: SiLU-gate (v_rcp_f32 fast path), write bf16 H rows into LDS
        #pragma unroll
        for (int nt = 0; nt < 8; ++nt) {
            #pragma unroll
            for (int v = 0; v < 8; ++v) {
                const float g = acc1[nt][v];
                const float u = acc3[nt][v];
                const float h = g * fast_rcp(1.0f + __expf(-g)) * u;
                const int m   = wave * 16 + v + hhi * 8;
                const int col = nc * BN + nt * 16 + lhalf;
                Hs[m][col] = f2bf(h);
            }
        }
    }

    // ================= Phase B: OUT = H * W2 (A straight from LDS) =========
    for (int nc = 0; nc < NCH; ++nc) {
        f32x8 acc[8] = {};
        const unsigned short* w2nc = w2te + (size_t)(nc * BN) * DIM;

        if (wave == 0) {
            tdm_load_wtile(w2nc, &WA[0][0][0]);
            __builtin_amdgcn_s_wait_tensorcnt(0);
        }
        for (int kb = 0; kb < NKB; ++kb) {
            const int cur = kb & 1, nxt = cur ^ 1;
            __syncthreads();                 // also fences Hs writes on first pass
            if (kb + 1 < NKB && wave == 0)
                tdm_load_wtile(w2nc + (kb + 1) * KB, &WA[nxt][0][0]);

            const bf16x16 a = frag_a(&Hs[wave * 16 + lhalf][kb * KB], hhi);
            #pragma unroll
            for (int nt = 0; nt < 8; ++nt) {
                const int n = nt * 16 + lhalf;
                acc[nt] = wmma_bf16(a, frag_b(&WA[cur][n][hhi * 16]), acc[nt]);
            }
            if (kb + 1 < NKB && wave == 0) __builtin_amdgcn_s_wait_tensorcnt(0);
        }
        #pragma unroll
        for (int nt = 0; nt < 8; ++nt) {
            #pragma unroll
            for (int v = 0; v < 8; ++v) {
                const int m = wave * 16 + v + hhi * 8;
                const long long gr = row0 + m;
                if (gr < rows) {
                    const int col = nc * BN + nt * 16 + lhalf;
                    oe[(size_t)gr * DIM + col] = acc[nt][v];
                }
            }
        }
    }
}

extern "C" void kernel_launch(void* const* d_in, const int* in_sizes, int n_in,
                              void* d_out, int out_size, void* d_ws, size_t ws_size,
                              hipStream_t stream) {
    (void)in_sizes; (void)n_in; (void)out_size; (void)ws_size;
    const float*     x  = (const float*)d_in[0];
    const float*     w1 = (const float*)d_in[1];
    const float*     w2 = (const float*)d_in[2];
    const float*     w3 = (const float*)d_in[3];
    const long long* ms = (const long long*)d_in[4];
    float*           o  = (float*)d_out;

    // d_ws layout: 24 matrices of 1024x1024 bf16, transposed to [n][k]:
    //   z 0..7 = w1t per expert, z 8..15 = w3t, z 16..23 = w2t  (48 MB total)
    unsigned short* wsT = (unsigned short*)d_ws;
    unsigned short* w1t = wsT;
    unsigned short* w3t = wsT + (size_t)8  * MEL;
    unsigned short* w2t = wsT + (size_t)16 * MEL;

    dim3 cgrid(16, 16, 24);
    moe_wcvt_kernel<<<cgrid, 256, 0, stream>>>(w1, w2, w3, wsT);

    dim3 grid(TOTAL_TOKENS / NEXP / BM, NEXP);   // (64, 8)
    moe_ffn_fused_kernel<<<grid, 256, 0, stream>>>(x, w1t, w3t, w2t, ms, o);
}
